// GRAPHEDX_Dual_xor_87694642250028
// MI455X (gfx1250) — matrix operand order repaired and checked by reading, not compiled, not run on hardware
//
#include <hip/hip_runtime.h>
#include <hip/hip_bf16.h>

typedef __attribute__((ext_vector_type(16))) _Float16 v16h;
typedef __attribute__((ext_vector_type(8)))  _Float16 v8h;
typedef __attribute__((ext_vector_type(8)))  float    v8f;

#define B_     48
#define NMAX_  20
#define NC_    16
#define D_     32
#define EH_    64
#define NTOT_  1728          // 48*(20+16)
#define NE_    6720          // 48*(80+60)
#define NPAIR_ 190           // triu(20,1)
#define TEMP_  0.01f

// f16 weight workspace offsets (in halves); each region is fragment-ordered:
// frag[kt][nt][lane(32)][half(16)], half i of lane (g= lane>>4, n= lane&15)
// holds W[kt*32 + 8*g + i + 8*(i>>3)][nt*16 + n] (zero-padded past source K; for msg/lrl
// layer-1 the bias is stored in pad row 95 and X row 95 is set to 1.0)
#define OFF_MW1 0            // msg_w1 96x64  (KT=3,NT=4)  bias in row 95
#define OFF_MW2 6144         // msg_w2 64x64  (KT=2,NT=4)
#define OFF_UW1 10240        // upd_w1 96x64  (KT=3,NT=4)
#define OFF_UW2 16384        // upd_w2 64x32  (KT=2,NT=2)
#define OFF_LW1 18432        // lrl_w1 96x64 padded (KT=3,NT=4) bias in row 95
#define OFF_LW2 24576        // lrl_w2 64x64  (KT=2,NT=4)
#define WH_TOT  28672

__device__ __forceinline__ void pair_sd(int p, int& s, int& d) {
  int row = 0, rem = p, cnt = NMAX_ - 1;
  while (rem >= cnt) { rem -= cnt; ++row; --cnt; }
  s = row; d = row + 1 + rem;
}

// ---------------- weight prep: f32 -> f16 fragments ----------------
__device__ __forceinline__ float frag_src(const float* W, const float* bias95,
                                          int f, int NT, int Ksrc, int ncols) {
  int i = f & 15, lane = (f >> 4) & 31, fr = f >> 9;
  int nt = fr % NT, kt = fr / NT;
  int k = kt * 32 + 8 * (lane >> 4) + i + ((i >> 3) << 3);
  int col = nt * 16 + (lane & 15);
  if (k < Ksrc) return W[k * ncols + col];
  if (bias95 != nullptr && k == 95) return bias95[col];
  return 0.f;
}

__global__ void k_prep(const float* mw1, const float* mb1, const float* mw2,
                       const float* uw1, const float* uw2,
                       const float* lw1, const float* lb1, const float* lw2,
                       _Float16* w) {
  int t = blockIdx.x * blockDim.x + threadIdx.x;
  if (t >= WH_TOT) return;
  int idx = t;
  float v;
  if (idx < 6144)                v = frag_src(mw1, mb1, idx, 4, 80, 64);
  else if ((idx -= 6144) < 4096) v = frag_src(mw2, nullptr, idx, 4, 64, 64);
  else if ((idx -= 4096) < 6144) v = frag_src(uw1, nullptr, idx, 4, 96, 64);
  else if ((idx -= 6144) < 2048) v = frag_src(uw2, nullptr, idx, 2, 64, 32);
  else if ((idx -= 2048) < 6144) v = frag_src(lw1, lb1, idx, 4, 65, 64);
  else { idx -= 6144;            v = frag_src(lw2, nullptr, idx, 4, 64, 64); }
  w[t] = (_Float16)v;
}

// ---------------- encoders ----------------
__global__ void k_encn(const float* nf, const float* w, const float* b, float* h) {
  int t = blockIdx.x * blockDim.x + threadIdx.x;
  if (t >= NTOT_ * D_) return;
  int n = t >> 5, d = t & 31;
  float s = b[d];
  #pragma unroll
  for (int k = 0; k < 16; ++k) s += nf[n * 16 + k] * w[k * 32 + d];
  h[t] = s;
}
__global__ void k_ence(const float* ef, const float* w, const float* b, float* e) {
  int t = blockIdx.x * blockDim.x + threadIdx.x;
  if (t >= NE_ * 16) return;
  int n = t >> 4, d = t & 15;
  float s = b[d];
  #pragma unroll
  for (int k = 0; k < 8; ++k) s += ef[n * 8 + k] * w[k * 16 + d];
  e[t] = s;
}

// ---------------- WMMA helpers ----------------
// convert 8 consecutive f32 -> 8 f16 (two float4 loads, packed cvt)
__device__ __forceinline__ v8h cvt8(const float* src) {
  float4 a = ((const float4*)src)[0];
  float4 b = ((const float4*)src)[1];
  v8h r;
  r[0] = (_Float16)a.x; r[1] = (_Float16)a.y; r[2] = (_Float16)a.z; r[3] = (_Float16)a.w;
  r[4] = (_Float16)b.x; r[5] = (_Float16)b.y; r[6] = (_Float16)b.z; r[7] = (_Float16)b.w;
  return r;
}
__device__ __forceinline__ v16h pack16(v8h lo, v8h hi) {
  union { v16h v; v8h h[2]; } u;
  u.h[0] = lo; u.h[1] = hi;
  return u.v;
}
__device__ __forceinline__ v8h zero8() {
  v8h r;
  #pragma unroll
  for (int i = 0; i < 8; ++i) r[i] = (_Float16)0.f;
  return r;
}

// one wave computes a 16-row MLP2 tile: Y2 = relu(X@W1[+b1])@W2+b2 ; X is 16x96 in A[3]
// W1/W2 fragment-ordered (see k_prep). C matrices start as inline-0; biases are
// folded into W1 row 95 (FOLD1) or added after the WMMA chain.
template <int NT2, bool FOLD1>
__device__ __forceinline__ void mlp2_wave(const v16h A[3],
    const _Float16* W1, const float* b1, const _Float16* W2, const float* b2,
    _Float16 (*Y1)[64], int lane, v8f* acc2) {
  int g = lane >> 4, n = lane & 15;
  const v16h* W1f = (const v16h*)W1;
  const v16h* W2f = (const v16h*)W2;
  #pragma unroll
  for (int nt = 0; nt < 4; ++nt) {
    v8f acc = {};                                  // SRC2 = inline 0
    #pragma unroll
    for (int kt = 0; kt < 3; ++kt) {
      v16h Bf = W1f[(kt * 4 + nt) * 32 + lane];
      acc = __builtin_amdgcn_wmma_f32_16x16x32_f16(false, A[kt], false, Bf,
                                                   (short)0, acc, false, false);
    }
    float bias;
    if constexpr (FOLD1) bias = 0.f; else bias = b1[nt * 16 + n];
    #pragma unroll
    for (int r = 0; r < 8; ++r) {                  // D: row M = r+8g, col nt*16+n
      float v;
      if constexpr (FOLD1) v = acc[r]; else v = acc[r] + bias;
      Y1[r + 8 * g][nt * 16 + n] = (_Float16)fmaxf(v, 0.f);
    }
  }
  __syncthreads();
  const v8h* yrow = (const v8h*)Y1[n];  // row m = n, 64 halves = 8 x v8h
  v16h A2[2];
  #pragma unroll
  for (int kt = 0; kt < 2; ++kt)        // halves 0..7 -> k=kt*32+8g.., 8..15 -> +16
    A2[kt] = pack16(yrow[kt * 4 + g], yrow[kt * 4 + 2 + g]);
  #pragma unroll
  for (int nt = 0; nt < NT2; ++nt) {
    v8f acc = {};                                  // SRC2 = inline 0
    #pragma unroll
    for (int kt = 0; kt < 2; ++kt) {
      v16h Bf = W2f[(kt * NT2 + nt) * 32 + lane];
      acc = __builtin_amdgcn_wmma_f32_16x16x32_f16(false, A2[kt], false, Bf,
                                                   (short)0, acc, false, false);
    }
    float bias = b2[nt * 16 + n];
    v8f o;
    #pragma unroll
    for (int r = 0; r < 8; ++r) o[r] = acc[r] + bias;
    acc2[nt] = o;
  }
}

// ---------------- message MLP: 6720 x ([h_from|h_to|e] 80) -> 64 -> 64 -------
__global__ __launch_bounds__(128) void k_msg(const float* h, const float* e,
    const int* from_idx, const int* to_idx, const _Float16* wh,
    const float* b2, float* msg) {
  __shared__ __align__(16) _Float16 Y1[4][16][64];
  int wave = threadIdx.x >> 5, lane = threadIdx.x & 31;
  int g = lane >> 4, n = lane & 15;
  int tile = blockIdx.x * 4 + wave;
  int row = tile * 16 + n;
  const float* hf = h + from_idx[row] * D_;
  const float* ht = h + to_idx[row] * D_;
  const float* ee = e + row * 16;
  v16h A[3];
  A[0] = pack16(cvt8(hf + 8 * g), cvt8(hf + 16 + 8 * g));
  A[1] = pack16(cvt8(ht + 8 * g), cvt8(ht + 16 + 8 * g));
  v8h hi = zero8();
  hi[7] = (_Float16)1.f;      // X[87+8g]: g=1 hits bias row 95; g=0 hits zero row
  A[2] = pack16(cvt8(ee + 8 * g), hi);
  v8f acc2[4];
  mlp2_wave<4, true>(A, wh + OFF_MW1, nullptr, wh + OFF_MW2, b2, Y1[wave], lane, acc2);
  #pragma unroll
  for (int nt = 0; nt < 4; ++nt)
    #pragma unroll
    for (int r = 0; r < 8; ++r)
      msg[(tile * 16 + r + 8 * g) * 64 + nt * 16 + n] = acc2[nt][r];
}

// ---------------- zero / scatter ----------------
__global__ void k_zero(float* p, int n) {
  int t = blockIdx.x * blockDim.x + threadIdx.x;
  if (t < n) p[t] = 0.f;
}
__global__ void k_scatter(const float* msg, const int* to_idx, float* agg) {
  int t = blockIdx.x * blockDim.x + threadIdx.x;
  if (t >= NE_ * EH_) return;
  int eid = t >> 6, c = t & 63;
  atomicAdd(&agg[to_idx[eid] * EH_ + c], msg[t]);
}

// ---------------- update MLP: 1728 x ([agg|h] 96) -> 64 -> 32, residual -----
__global__ __launch_bounds__(128) void k_upd(const float* agg, const _Float16* wh,
    const float* b1, const float* b2, float* h) {
  __shared__ __align__(16) _Float16 Y1[4][16][64];
  int wave = threadIdx.x >> 5, lane = threadIdx.x & 31;
  int g = lane >> 4, n = lane & 15;
  int tile = blockIdx.x * 4 + wave;
  int row = tile * 16 + n;
  const float* xa = agg + row * EH_;
  const float* xh = h + row * D_;
  v16h A[3];
  A[0] = pack16(cvt8(xa + 8 * g),      cvt8(xa + 16 + 8 * g));
  A[1] = pack16(cvt8(xa + 32 + 8 * g), cvt8(xa + 48 + 8 * g));
  A[2] = pack16(cvt8(xh + 8 * g),      cvt8(xh + 16 + 8 * g));
  v8f acc2[2];
  mlp2_wave<2, false>(A, wh + OFF_UW1, b1, wh + OFF_UW2, b2, Y1[wave], lane, acc2);
  #pragma unroll
  for (int nt = 0; nt < 2; ++nt)
    #pragma unroll
    for (int r = 0; r < 8; ++r)
      h[(tile * 16 + r + 8 * g) * D_ + nt * 16 + n] += acc2[nt][r];
}

// ---------------- split h -> Hq (48x20x32), Hc padded (48x20x32) ----------
__global__ void k_split(const float* h, float* Hq, float* Hc) {
  int t = blockIdx.x * blockDim.x + threadIdx.x;        // in float4 units
  if (t >= B_ * 20 * 8) return;
  int b = t / 160, rem = t - b * 160;
  int i = rem >> 3, k4 = rem & 7;
  const float4* h4 = (const float4*)h;
  float4 z; z.x = z.y = z.z = z.w = 0.f;
  ((float4*)Hq)[t] = h4[(b * 36 + i) * 8 + k4];
  ((float4*)Hc)[t] = (i < NC_) ? h4[(b * 36 + 20 + i) * 8 + k4] : z;
}

// ---------------- sink MLP -> tq, tc (48x20x20) ----------------
__global__ void k_sink(const float* Hq, const float* Hc, const float* w1, const float* b1,
                       const float* w2, const float* b2, float* tq, float* tc) {
  int t = blockIdx.x * blockDim.x + threadIdx.x;
  if (t >= 2 * B_ * 20) return;
  int which = t / (B_ * 20), r = t - which * (B_ * 20);
  int b = r / 20, i = r - b * 20;
  const float* H = (which ? Hc : Hq) + (b * 20 + i) * D_;
  float hid[20];
  #pragma unroll
  for (int j = 0; j < 20; ++j) {
    float s = b1[j];
    #pragma unroll
    for (int k = 0; k < 32; ++k) s += H[k] * w1[k * 20 + j];
    hid[j] = fmaxf(s, 0.f);
  }
  float* T = (which ? tc : tq) + b * 400 + i * 20;
  #pragma unroll
  for (int j2 = 0; j2 < 20; ++j2) {
    float s = b2[j2];
    #pragma unroll
    for (int j = 0; j < 20; ++j) s += hid[j] * w2[j * 20 + j2];
    T[j2] = s;
  }
}

// ---------------- L1 cost + Sinkhorn (20 iters), zero the output ----------
__global__ void k_sinkhorn(const float* tq, const float* tc, float* P, float* out) {
  int b = blockIdx.x;
  int j = threadIdx.x, i = threadIdx.y;
  __shared__ float TQ[20][20], TC[20][20], LA[20][20], LR[20], LC[20];
  TQ[i][j] = tq[b * 400 + i * 20 + j];
  TC[i][j] = tc[b * 400 + i * 20 + j];
  __syncthreads();
  float c = 0.f;
  #pragma unroll
  for (int k = 0; k < 20; ++k) c += fabsf(TQ[i][k] - TC[j][k]);
  LA[i][j] = -c * (1.0f / TEMP_);
  for (int it = 0; it < 20; ++it) {
    __syncthreads();
    if (j == 0) {                       // axis 2 (over j)
      float m = -3.0e38f;
      for (int jj = 0; jj < 20; ++jj) m = fmaxf(m, LA[i][jj]);
      float s = 0.f;
      for (int jj = 0; jj < 20; ++jj) s += expf(LA[i][jj] - m);
      LR[i] = m + logf(s);
    }
    __syncthreads();
    LA[i][j] -= LR[i];
    __syncthreads();
    if (i == 0) {                       // axis 1 (over i)
      float m = -3.0e38f;
      for (int ii = 0; ii < 20; ++ii) m = fmaxf(m, LA[ii][j]);
      float s = 0.f;
      for (int ii = 0; ii < 20; ++ii) s += expf(LA[ii][j] - m);
      LC[j] = m + logf(s);
    }
    __syncthreads();
    LA[i][j] -= LC[j];
  }
  P[b * 400 + i * 20 + j] = expf(LA[i][j]);
  if (i == 0 && j == 0) out[b] = 0.f;
}

// ---------------- edge embeddings: 2 graphs x 48 x 190 x 2 dirs = 36480 rows -
__global__ __launch_bounds__(128) void k_lrl(const float* Hq, const float* Hc,
    const float* qadj, const float* cadj, const _Float16* wh,
    const float* b2, float* Eq, float* Ec) {
  __shared__ __align__(16) _Float16 Y1[4][16][64];
  int wave = threadIdx.x >> 5, lane = threadIdx.x & 31;
  int g = lane >> 4, n = lane & 15;
  int tile = blockIdx.x * 4 + wave;
  int rowid = tile * 16 + n;
  int dir = rowid & 1;
  int t2 = rowid >> 1;
  int p = t2 % NPAIR_;
  int t3 = t2 / NPAIR_;
  int b = t3 % B_;
  int gsel = t3 / B_;
  int s, d; pair_sd(p, s, d);
  const float* H = (gsel ? Hc : Hq) + b * 20 * D_;
  const float* AD = (gsel ? cadj : qadj) + b * 400;
  float aval = AD[s * 20 + d];
  int n1 = dir ? d : s, n2 = dir ? s : d;
  const float* x1 = H + n1 * D_;
  const float* x2 = H + n2 * D_;
  v16h A[3];
  A[0] = pack16(cvt8(x1 + 8 * g), cvt8(x1 + 16 + 8 * g));
  A[1] = pack16(cvt8(x2 + 8 * g), cvt8(x2 + 16 + 8 * g));
  v16h a2;
  #pragma unroll
  for (int i = 0; i < 16; ++i) a2[i] = (_Float16)0.f;
  a2[0]  = (_Float16)aval;   // X[64+8g]: g=0 -> adj slot; g=1 aliases zero row 72
  a2[15] = (_Float16)1.f;    // X[87+8g]: g=1 -> bias row 95; g=0 aliases zero row 87
  A[2] = a2;
  v8f acc2[4];
  mlp2_wave<4, true>(A, wh + OFF_LW1, nullptr, wh + OFF_LW2, b2, Y1[wave], lane, acc2);
  #pragma unroll
  for (int nt = 0; nt < 4; ++nt)
    #pragma unroll
    for (int r = 0; r < 8; ++r) {
      int rr = tile * 16 + r + 8 * g;
      int tt2 = rr >> 1;
      int pp = tt2 % NPAIR_;
      int tt3 = tt2 / NPAIR_;
      int bb = tt3 % B_;
      int gg = tt3 / B_;
      float* E = gg ? Ec : Eq;
      atomicAdd(&E[(bb * NPAIR_ + pp) * 64 + nt * 16 + n], acc2[nt][r]);  // fwd + bk
    }
}

// ---------------- edge alignment reduction ----------------
__global__ void k_edge(const float* Eq, const float* Ec, const float* P,
                       const float* qadj, const float* cadj, float* out) {
  __shared__ __align__(16) float sEc[NPAIR_ * 64];
  __shared__ float sP[400];
  __shared__ int sS[NPAIR_], sD[NPAIR_], sAq[NPAIR_], sAc[NPAIR_];
  __shared__ float red[256];
  int b = blockIdx.x, tid = threadIdx.x;
  const float4* gEc = (const float4*)(Ec + b * NPAIR_ * 64);
  for (int i = tid; i < NPAIR_ * 16; i += 256) ((float4*)sEc)[i] = gEc[i];
  for (int i = tid; i < 400; i += 256) sP[i] = P[b * 400 + i];
  for (int i = tid; i < NPAIR_; i += 256) {
    int s, d; pair_sd(i, s, d);
    sS[i] = s; sD[i] = d;
    sAq[i] = (qadj[b * 400 + s * 20 + d] != 0.f);
    sAc[i] = (cadj[b * 400 + s * 20 + d] != 0.f);
  }
  __syncthreads();
  float accum = 0.f;
  if (tid < NPAIR_) {
    int e1 = tid;
    float4 q4[16];
    const float4* gq = (const float4*)(Eq + (b * NPAIR_ + e1) * 64);
    #pragma unroll
    for (int k = 0; k < 16; ++k) q4[k] = gq[k];
    int s1 = sS[e1], d1 = sD[e1], a1 = sAq[e1];
    const float* Ps1 = sP + s1 * 20;
    const float* Pd1 = sP + d1 * 20;
    for (int e2 = 0; e2 < NPAIR_; ++e2) {
      if (a1 == sAc[e2]) continue;      // pxor == 0
      int s2 = sS[e2], d2 = sD[e2];
      float eT = Ps1[s2] * Pd1[d2] + Ps1[d2] * Pd1[s2];
      const float4* c4 = (const float4*)(sEc + e2 * 64);
      float ec = 0.f;
      #pragma unroll
      for (int k = 0; k < 16; ++k) {
        float4 cc = c4[k];
        ec += fabsf(q4[k].x - cc.x) + fabsf(q4[k].y - cc.y) +
              fabsf(q4[k].z - cc.z) + fabsf(q4[k].w - cc.w);
      }
      accum += eT * ec;
    }
  }
  red[tid] = accum;
  __syncthreads();
  for (int s = 128; s > 0; s >>= 1) {
    if (tid < s) red[tid] += red[tid + s];
    __syncthreads();
  }
  if (tid == 0) atomicAdd(out + b, red[0]);
}

// ---------------- node alignment reduction ----------------
__global__ void k_node(const float* Hq, const float* Hc, const float* P, float* out) {
  __shared__ float red[512];
  int b = blockIdx.x, tid = threadIdx.x;
  float acc = 0.f;
  if (tid < 400) {
    int i = tid / 20, j = tid - (tid / 20) * 20;
    int nxor = 1 ^ (j < NC_ ? 1 : 0);   // NXOR = indq ^ indc
    if (nxor) {
      const float4* hq4 = (const float4*)(Hq + (b * 20 + i) * D_);
      const float4* hc4 = (const float4*)(Hc + (b * 20 + j) * D_);
      float nc = 0.f;
      #pragma unroll
      for (int k = 0; k < 8; ++k) {
        float4 a = hq4[k], c = hc4[k];
        nc += fabsf(a.x - c.x) + fabsf(a.y - c.y) + fabsf(a.z - c.z) + fabsf(a.w - c.w);
      }
      acc = P[b * 400 + i * 20 + j] * nc;
    }
  }
  red[tid] = acc;
  __syncthreads();
  for (int s = 256; s > 0; s >>= 1) {
    if (tid < s) red[tid] += red[tid + s];
    __syncthreads();
  }
  if (tid == 0) atomicAdd(out + b, red[0]);
}

// ---------------- launcher ----------------
extern "C" void kernel_launch(void* const* d_in, const int* in_sizes, int n_in,
                              void* d_out, int out_size, void* d_ws, size_t ws_size,
                              hipStream_t stream) {
  (void)in_sizes; (void)n_in; (void)out_size; (void)ws_size;
  const float* node_features = (const float*)d_in[0];
  const float* edge_features = (const float*)d_in[1];
  const float* qadj   = (const float*)d_in[2];
  const float* cadj   = (const float*)d_in[3];
  const float* enc_nw = (const float*)d_in[4];
  const float* enc_nb = (const float*)d_in[5];
  const float* enc_ew = (const float*)d_in[6];
  const float* enc_eb = (const float*)d_in[7];
  const float* msg_w1 = (const float*)d_in[8];
  const float* msg_b1 = (const float*)d_in[9];
  const float* msg_w2 = (const float*)d_in[10];
  const float* msg_b2 = (const float*)d_in[11];
  const float* upd_w1 = (const float*)d_in[12];
  const float* upd_b1 = (const float*)d_in[13];
  const float* upd_w2 = (const float*)d_in[14];
  const float* upd_b2 = (const float*)d_in[15];
  const float* sink_w1 = (const float*)d_in[16];
  const float* sink_b1 = (const float*)d_in[17];
  const float* sink_w2 = (const float*)d_in[18];
  const float* sink_b2 = (const float*)d_in[19];
  const float* lrl_w1 = (const float*)d_in[20];
  const float* lrl_b1 = (const float*)d_in[21];
  const float* lrl_w2 = (const float*)d_in[22];
  const float* lrl_b2 = (const float*)d_in[23];
  const int* from_idx = (const int*)d_in[24];
  const int* to_idx   = (const int*)d_in[25];
  float* out = (float*)d_out;

  float* ws = (float*)d_ws;
  float* h    = ws; ws += NTOT_ * D_;
  float* eenc = ws; ws += NE_ * 16;
  float* msg  = ws; ws += NE_ * EH_;
  float* agg  = ws; ws += NTOT_ * EH_;
  float* Hq   = ws; ws += B_ * 20 * D_;
  float* Hc   = ws; ws += B_ * 20 * D_;
  float* tq   = ws; ws += B_ * 400;
  float* tc   = ws; ws += B_ * 400;
  float* P    = ws; ws += B_ * 400;
  float* Eq   = ws; ws += B_ * NPAIR_ * 64;
  float* Ec   = ws; ws += B_ * NPAIR_ * 64;
  _Float16* wh = (_Float16*)ws;

  k_prep<<<(WH_TOT + 255) / 256, 256, 0, stream>>>(msg_w1, msg_b1, msg_w2,
                                                   upd_w1, upd_w2,
                                                   lrl_w1, lrl_b1, lrl_w2, wh);
  k_encn<<<(NTOT_ * D_ + 255) / 256, 256, 0, stream>>>(node_features, enc_nw, enc_nb, h);
  k_ence<<<(NE_ * 16 + 255) / 256, 256, 0, stream>>>(edge_features, enc_ew, enc_eb, eenc);

  for (int it = 0; it < 5; ++it) {
    k_msg<<<NE_ / 64, 128, 0, stream>>>(h, eenc, from_idx, to_idx, wh, msg_b2, msg);
    k_zero<<<(NTOT_ * EH_ + 255) / 256, 256, 0, stream>>>(agg, NTOT_ * EH_);
    k_scatter<<<(NE_ * EH_ + 255) / 256, 256, 0, stream>>>(msg, to_idx, agg);
    k_upd<<<NTOT_ / 64, 128, 0, stream>>>(agg, wh, upd_b1, upd_b2, h);
  }

  k_split<<<(B_ * 20 * 8 + 255) / 256, 256, 0, stream>>>(h, Hq, Hc);
  k_sink<<<(2 * B_ * 20 + 127) / 128, 128, 0, stream>>>(Hq, Hc, sink_w1, sink_b1,
                                                        sink_w2, sink_b2, tq, tc);
  k_sinkhorn<<<B_, dim3(20, 20), 0, stream>>>(tq, tc, P, out);
  k_zero<<<(2 * B_ * NPAIR_ * 64 + 255) / 256, 256, 0, stream>>>(Eq, 2 * B_ * NPAIR_ * 64);
  k_lrl<<<(2 * B_ * NPAIR_ * 2) / 64, 128, 0, stream>>>(Hq, Hc, qadj, cadj, wh,
                                                        lrl_b2, Eq, Ec);
  k_edge<<<B_, 256, 0, stream>>>(Eq, Ec, P, qadj, cadj, out);
  k_node<<<B_, 512, 0, stream>>>(Hq, Hc, P, out);
}
